// HybridEngine_61512521613861
// MI455X (gfx1250) — compile-verified, matrix operand order checked by reading, not run on hardware
//
#include <hip/hip_runtime.h>
#include <hip/hip_bf16.h>
#include <math.h>

#define BQ 2
#define SQ 1024
#define DQ 1024
#define VQ 50257
#define NCQ 28
#define NSQ 4
#define NPQ 32
#define NMQ 4
#define PDQ 32
#define EPSQ 1.1920929e-07f
#define ROWS (BQ*SQ)   // 2048
#define GSTR 40        // LDS row stride in bf16 (64B row + 16B TDM pad) -> conflict-free

typedef __attribute__((ext_vector_type(16))) __bf16 v16bf;
typedef __attribute__((ext_vector_type(8)))  float  v8f;
typedef __attribute__((ext_vector_type(4)))  unsigned int v4u;
typedef __attribute__((ext_vector_type(8)))  unsigned int v8u;

union Frag { v16bf v; uint4 q[2]; };

__device__ __forceinline__ v8f wmma_bf16(const v16bf& a, const v16bf& b, const v8f& c) {
  // D = A(16x32 bf16) x B(32x16 bf16) + C(16x16 f32)
  return __builtin_amdgcn_wmma_f32_16x16x32_bf16(false, a, false, b, (short)0, c, false, false);
}

// ---------------- embed: x[row,d] = embed_w[tokens[row], d] ----------------
__global__ void k_embed(const int* __restrict__ tok, const float* __restrict__ ew,
                        float* __restrict__ x) {
  int i = blockIdx.x * 256 + threadIdx.x;      // ROWS*DQ threads
  int row = i >> 10, d = i & 1023;
  x[i] = ew[(size_t)tok[row] * DQ + d];
}

// ---------------- rotation on NP planes (xn pre-copied from x) -------------
__global__ void k_rotate(const float* __restrict__ x, float* __restrict__ xn,
                         const float* __restrict__ ang, const int* __restrict__ pi_,
                         const int* __restrict__ pj_) {
  int i = blockIdx.x * 256 + threadIdx.x;      // ROWS*NPQ threads
  int k = i & (NPQ - 1); int row = i >> 5;
  float a = ang[k], ca = cosf(a), sa = sinf(a);
  int ii = pi_[k], jj = pj_[k];
  size_t base = (size_t)row * DQ;
  float xi = x[base + ii], xj = x[base + jj];
  xn[base + ii] = xi * ca - xj * sa;
  xn[base + jj] = xi * sa + xj * ca;
}

// ---------------- column mean over S:  m[b,d] ------------------------------
__global__ void k_colmean(const float* __restrict__ xn, float* __restrict__ m) {
  int t = blockIdx.x * 256 + threadIdx.x;      // BQ*DQ threads
  int b = t >> 10, d = t & 1023;
  const float* p = xn + (size_t)b * SQ * DQ + d;
  float s = 0.f;
  for (int i = 0; i < SQ; ++i) s += p[(size_t)i * DQ];
  m[t] = s * (1.0f / SQ);
}

// ---------------- router softmax -> gamma/beta per batch -------------------
__global__ void k_router(const float* __restrict__ m, const float* __restrict__ rw,
                         const float* __restrict__ msc, const float* __restrict__ msh,
                         float* __restrict__ gamma, float* __restrict__ beta) {
  int b = blockIdx.x, tid = threadIdx.x;
  __shared__ float sm[NMQ][256];
  __shared__ float wgt[NMQ];
  float acc[NMQ] = {0.f, 0.f, 0.f, 0.f};
  for (int d = tid; d < DQ; d += 256) {
    float md = m[b * DQ + d];
    #pragma unroll
    for (int nm = 0; nm < NMQ; ++nm) acc[nm] += md * rw[nm * DQ + d];
  }
  #pragma unroll
  for (int nm = 0; nm < NMQ; ++nm) sm[nm][tid] = acc[nm];
  __syncthreads();
  for (int off = 128; off > 0; off >>= 1) {
    if (tid < off) {
      #pragma unroll
      for (int nm = 0; nm < NMQ; ++nm) sm[nm][tid] += sm[nm][tid + off];
    }
    __syncthreads();
  }
  if (tid == 0) {
    float mx = sm[0][0];
    for (int nm = 1; nm < NMQ; ++nm) mx = fmaxf(mx, sm[nm][0]);
    float e[NMQ], ssum = 0.f;
    for (int nm = 0; nm < NMQ; ++nm) { e[nm] = expf(sm[nm][0] - mx); ssum += e[nm]; }
    for (int nm = 0; nm < NMQ; ++nm) wgt[nm] = e[nm] / ssum;
  }
  __syncthreads();
  for (int d = tid; d < DQ; d += 256) {
    float g = 0.f, bb = 0.f;
    #pragma unroll
    for (int nm = 0; nm < NMQ; ++nm) { g += wgt[nm] * msc[nm * DQ + d]; bb += wgt[nm] * msh[nm * DQ + d]; }
    gamma[b * DQ + d] = g; beta[b * DQ + d] = bb;
  }
}

// ---------------- f32 -> bf16 flat convert ---------------------------------
__global__ void k_cvt_bf16(const float* __restrict__ a, __bf16* __restrict__ o, long long n) {
  long long i = (long long)blockIdx.x * 256 + threadIdx.x;
  if (i < n) o[i] = (__bf16)a[i];
}

// ---------------- fused cycle MLP + rmsnorm residual -----------------------
// One wave = 16-row tile.  GEMM1: (16x1024)x(1024x32) ; GEMM2: (16x32)x(32x1024)
__global__ void k_cycle_mlp(float* __restrict__ x, const float* __restrict__ xn,
                            const float* __restrict__ gamma, const float* __restrict__ beta,
                            const __bf16* __restrict__ w1b, const float* __restrict__ b1,
                            const __bf16* __restrict__ w2b, const float* __restrict__ b2,
                            const float* __restrict__ nw, const float* __restrict__ cyc_scale,
                            int ci, float* __restrict__ dlt) {
  __shared__ __align__(16) float hls[8][16][32];   // 16 KB: h tile bounce per wave
  int wave = threadIdx.x >> 5;
  int lane = threadIdx.x & 31;
  int row0 = blockIdx.x * 128 + wave * 16;
  int mloc = lane & 15;
  int ko   = (lane < 16) ? 0 : 8;                  // K-half split per ISA A layout
  int row  = row0 + mloc;
  int b    = row >> 10;
  const float* xr = xn + (size_t)row * DQ;
  const float* gr = gamma + b * DQ;
  const float* br = beta  + b * DQ;

  v8f zc = {};
  v8f acc0 = zc, acc1 = zc;
  for (int k0 = 0; k0 < DQ; k0 += 32) {
    Frag af;
    #pragma unroll
    for (int t = 0; t < 16; ++t) {
      int kd = k0 + ko + t + ((t >= 8) ? 8 : 0);   // {0..7,16..23} / {8..15,24..31}
      af.v[t] = (__bf16)(xr[kd] * gr[kd] + br[kd]);
    }
    Frag bf0, bf1;
    bf0.q[0] = *(const uint4*)(w1b + (size_t)mloc * DQ + k0 + ko);
    bf0.q[1] = *(const uint4*)(w1b + (size_t)mloc * DQ + k0 + ko + 16);
    bf1.q[0] = *(const uint4*)(w1b + (size_t)(mloc + 16) * DQ + k0 + ko);
    bf1.q[1] = *(const uint4*)(w1b + (size_t)(mloc + 16) * DQ + k0 + ko + 16);
    acc0 = wmma_bf16(af.v, bf0.v, acc0);
    acc1 = wmma_bf16(af.v, bf1.v, acc1);
  }
  // h = exact GELU(acc + b1) -> LDS re-layout
  {
    int n = mloc;
    int mrow = (lane < 16) ? 0 : 8;                // C-layout rows per lane half
    #pragma unroll
    for (int r = 0; r < 8; ++r) {
      float v0 = acc0[r] + b1[n];
      float v1 = acc1[r] + b1[n + 16];
      v0 = 0.5f * v0 * (1.0f + erff(v0 * 0.70710678118654752f));
      v1 = 0.5f * v1 * (1.0f + erff(v1 * 0.70710678118654752f));
      hls[wave][mrow + r][n]      = v0;
      hls[wave][mrow + r][n + 16] = v1;
    }
  }
  __syncthreads();
  Frag a2;
  #pragma unroll
  for (int t = 0; t < 16; ++t) {
    int kd = ko + t + ((t >= 8) ? 8 : 0);
    a2.v[t] = (__bf16)hls[wave][mloc][kd];
  }
  float cs = 0.5f * cyc_scale[ci];
  float ssq[8];
  #pragma unroll
  for (int r = 0; r < 8; ++r) ssq[r] = 0.f;
  int mbase = (lane < 16) ? 0 : 8;
  for (int nt = 0; nt < DQ / 16; ++nt) {
    int n = nt * 16 + mloc;
    Frag bf2;
    bf2.q[0] = *(const uint4*)(w2b + (size_t)n * PDQ + ko);
    bf2.q[1] = *(const uint4*)(w2b + (size_t)n * PDQ + ko + 16);
    v8f c = wmma_bf16(a2.v, bf2.v, zc);
    #pragma unroll
    for (int r = 0; r < 8; ++r) {
      int rg = row0 + mbase + r;
      float v = c[r] + b2[n];
      v = v / (1.0f + expf(-v));                   // SiLU
      float d = v - x[(size_t)rg * DQ + n];
      dlt[(size_t)rg * DQ + n] = d;
      ssq[r] += d * d;
    }
  }
  // per-row sum of squares: reduce across the 16 lanes of each half-wave
  #pragma unroll
  for (int r = 0; r < 8; ++r) {
    float v = ssq[r];
    v += __shfl_xor(v, 1, 32);
    v += __shfl_xor(v, 2, 32);
    v += __shfl_xor(v, 4, 32);
    v += __shfl_xor(v, 8, 32);
    ssq[r] = v;
  }
  float rmsv[8];
  #pragma unroll
  for (int r = 0; r < 8; ++r) rmsv[r] = rsqrtf(ssq[r] * (1.0f / DQ) + EPSQ);
  for (int nt = 0; nt < DQ / 16; ++nt) {
    int n = nt * 16 + mloc;
    #pragma unroll
    for (int r = 0; r < 8; ++r) {
      int rg = row0 + mbase + r;
      float d = dlt[(size_t)rg * DQ + n];
      x[(size_t)rg * DQ + n] += cs * d * rmsv[r] * nw[n];
    }
  }
}

// ---------------- EMA scan over S ------------------------------------------
__global__ void k_ema(const float* __restrict__ x, const float* __restrict__ sd,
                      float* __restrict__ causal) {
  int t = blockIdx.x * 256 + threadIdx.x;         // BQ*DQ threads
  int b = t >> 10, d = t & 1023;
  float dec = 1.0f / (1.0f + expf(-sd[0]));
  float st = 0.f;
  const float* p = x + (size_t)b * SQ * DQ + d;
  float* q = causal + (size_t)b * SQ * DQ + d;
  for (int s = 0; s < SQ; ++s) {
    st = dec * st + (1.0f - dec) * p[(size_t)s * DQ];
    q[(size_t)s * DQ] = st;
  }
}

// ---------------- generic bf16 WMMA GEMM:  A(MxK) @ W(NxK)^T ---------------
// 128x128 block tile, 8 waves (2M x 4N), double-buffered LDS.
// B tile (128 rows x 64B, row stride K*2) streamed by the Tensor Data Mover:
// one tensor_load_to_lds per K-step with LDS pad -> 80B row stride
// (conflict-free fragment reads); TENSORcnt synchronized by wave 0 + barrier.
// TDM OOB rows (n >= N) are written as zeros, handling the V-tail with EXEC
// kept all-ones around WMMA.
// mode 0: out[m*DQ+n] += acc  (residual accumulate, N==DQ)
// mode 1: out[m*N +n]  = acc  (n<N predicated store)
__global__ void k_gemm(const __bf16* __restrict__ A, const __bf16* __restrict__ Wt,
                       float* __restrict__ out, int M, int N, int K, int mode) {
  __shared__ __align__(16) __bf16 Asm[2][128][GSTR];
  __shared__ __align__(16) __bf16 Bsm[2][128][GSTR];
  int tid  = threadIdx.x;
  int wave = tid >> 5, lane = tid & 31;
  int wm = wave & 1, wn = wave >> 1;              // 2x4 wave grid over 128x128 tile
  int m0 = blockIdx.y * 128;
  int n0 = blockIdx.x * 128;
  int mloc = lane & 15;
  int ko   = (lane < 16) ? 0 : 8;
  v8f zc = {};
  v8f acc[4][2];
  #pragma unroll
  for (int mi = 0; mi < 4; ++mi)
    #pragma unroll
    for (int ni = 0; ni < 2; ++ni) acc[mi][ni] = zc;

  auto loadA = [&](int buf, int k0) {             // 128x32 tile, 16 bf16 per thread
    int r  = tid >> 1;
    int cc = (tid & 1) * 16;
    const __bf16* src = A + (size_t)(m0 + r) * K + k0 + cc;
    *(uint4*)&Asm[buf][r][cc]     = *(const uint4*)(src);
    *(uint4*)&Asm[buf][r][cc + 8] = *(const uint4*)(src + 8);
  };

  auto issueTDM = [&](int buf, int k0) {          // D# per CDNA5 ISA ch.8 (2-group, 2D)
    unsigned lds = (unsigned)(size_t)(&Bsm[buf][0][0]);
    unsigned long long ga = (unsigned long long)(size_t)(Wt + (size_t)n0 * K + k0);
    v4u g0;
    g0[0] = 1u;                                             // count=1, no gather
    g0[1] = lds;                                            // lds_addr (bytes)
    g0[2] = (unsigned)(ga & 0xffffffffull);                 // global_addr[31:0]
    g0[3] = (unsigned)((ga >> 32) & 0x01ffffffull) | (2u << 30); // ga[56:32] | type=2
    unsigned long long td0 = (unsigned long long)(K - k0);  // remaining K extent
    unsigned long long td1 = (unsigned long long)(N - n0);  // remaining rows -> OOB=0
    unsigned long long q0 = 0ull, q1 = 0ull, q2 = 0ull, q3 = 0ull;
    q0 |= (1ull << 16);                    // data_size = 2 bytes
    q0 |= (1ull << 20);                    // pad_enable
    q0 |= (3ull << 22);                    // pad_interval: every 16 DWORDs (64B row)
    q0 |= (3ull << 25);                    // pad_amount:   4 DWORDs (16B) -> 80B stride
    q0 |= (td0 & 0xffffull) << 48;         // tensor_dim0[15:0]
    q1 |= (td0 >> 16) & 0xffffull;         // tensor_dim0[31:16]
    q1 |= (td1 & 0xffffffffull) << 16;     // tensor_dim1
    q1 |= 32ull << 48;                     // tile_dim0 = 32 elements
    q2 |= 128ull;                          // tile_dim1 = 128 rows
    q2 |= ((unsigned long long)(unsigned)K) << 32;  // tensor_dim0_stride[31:0] = K
    v8u g1;
    g1[0] = (unsigned)q0; g1[1] = (unsigned)(q0 >> 32);
    g1[2] = (unsigned)q1; g1[3] = (unsigned)(q1 >> 32);
    g1[4] = (unsigned)q2; g1[5] = (unsigned)(q2 >> 32);
    g1[6] = (unsigned)q3; g1[7] = (unsigned)(q3 >> 32);
    asm volatile("tensor_load_to_lds %0, %1" :: "s"(g0), "s"(g1) : "memory");
  };

  // preload buffer 0
  loadA(0, 0);
  if (wave == 0) {
    issueTDM(0, 0);
    __builtin_amdgcn_s_wait_tensorcnt(0);
  }
  __syncthreads();

  int nk = K / 32;
  for (int ki = 0; ki < nk; ++ki) {
    int cur = ki & 1;
    if (ki + 1 < nk) {                    // prefetch next tiles into other buffer
      loadA(cur ^ 1, (ki + 1) * 32);
      if (wave == 0) issueTDM(cur ^ 1, (ki + 1) * 32);
    }
    Frag bfr[2];
    #pragma unroll
    for (int ni = 0; ni < 2; ++ni) {
      int nrow = wn * 32 + ni * 16 + mloc;
      bfr[ni].q[0] = *(const uint4*)&Bsm[cur][nrow][ko];
      bfr[ni].q[1] = *(const uint4*)&Bsm[cur][nrow][ko + 16];
    }
    #pragma unroll
    for (int mi = 0; mi < 4; ++mi) {
      Frag af;
      int arow = wm * 64 + mi * 16 + mloc;
      af.q[0] = *(const uint4*)&Asm[cur][arow][ko];
      af.q[1] = *(const uint4*)&Asm[cur][arow][ko + 16];
      acc[mi][0] = wmma_bf16(af.v, bfr[0].v, acc[mi][0]);
      acc[mi][1] = wmma_bf16(af.v, bfr[1].v, acc[mi][1]);
    }
    if (wave == 0) __builtin_amdgcn_s_wait_tensorcnt(0);
    __syncthreads();
  }

  int mrow0 = m0 + wm * 64 + ((lane < 16) ? 0 : 8);
  #pragma unroll
  for (int mi = 0; mi < 4; ++mi) {
    #pragma unroll
    for (int ni = 0; ni < 2; ++ni) {
      int n = n0 + wn * 32 + ni * 16 + mloc;
      #pragma unroll
      for (int r = 0; r < 8; ++r) {
        int mm = mrow0 + mi * 16 + r;
        if (mode == 0) {
          out[(size_t)mm * DQ + n] += acc[mi][ni][r];
        } else if (n < N) {
          out[(size_t)mm * N + n] = acc[mi][ni][r];
        }
      }
    }
  }
}

// ---------------- final rmsnorm -> bf16 ------------------------------------
__global__ void k_finalnorm(const float* __restrict__ x, const float* __restrict__ onw,
                            __bf16* __restrict__ xfb) {
  int row = blockIdx.x, tid = threadIdx.x;
  __shared__ float sm[256];
  const float* p = x + (size_t)row * DQ;
  float s = 0.f;
  for (int d = tid; d < DQ; d += 256) { float v = p[d]; s += v * v; }
  sm[tid] = s; __syncthreads();
  for (int off = 128; off > 0; off >>= 1) {
    if (tid < off) sm[tid] += sm[tid + off];
    __syncthreads();
  }
  float rms = rsqrtf(sm[0] * (1.0f / DQ) + EPSQ);
  for (int d = tid; d < DQ; d += 256) xfb[(size_t)row * DQ + d] = (__bf16)(p[d] * rms * onw[d]);
}

// ===========================================================================
extern "C" void kernel_launch(void* const* d_in, const int* in_sizes, int n_in,
                              void* d_out, int out_size, void* d_ws, size_t ws_size,
                              hipStream_t stream) {
  (void)in_sizes; (void)n_in; (void)out_size; (void)ws_size;
  const int*   tokens  = (const int*)  d_in[0];
  const float* embed_w = (const float*)d_in[1];
  const float* angles  = (const float*)d_in[2];
  const int*   plane_i = (const int*)  d_in[3];
  const int*   plane_j = (const int*)  d_in[4];
  const float* msc     = (const float*)d_in[5];
  const float* msh     = (const float*)d_in[6];
  const float* rw      = (const float*)d_in[7];
  const float* w1      = (const float*)d_in[8];
  const float* b1      = (const float*)d_in[9];
  const float* w2      = (const float*)d_in[10];
  const float* b2      = (const float*)d_in[11];
  const float* nw      = (const float*)d_in[12];
  const float* sd      = (const float*)d_in[13];
  const float* sg      = (const float*)d_in[14];
  const float* csc     = (const float*)d_in[15];
  const float* onw     = (const float*)d_in[16];
  const float* lm      = (const float*)d_in[17];
  float* out = (float*)d_out;

  char* wp = (char*)d_ws;
  size_t off = 0;
  auto carve = [&](size_t bytes) -> void* {
    void* p = wp + off;
    off = (off + bytes + 255) & ~(size_t)255;
    return p;
  };
  float* x      = (float*) carve((size_t)ROWS * DQ * 4);
  float* xn     = (float*) carve((size_t)ROWS * DQ * 4);
  float* dlt    = (float*) carve((size_t)ROWS * DQ * 4);
  float* causal = (float*) carve((size_t)ROWS * DQ * 4);
  float* mbuf   = (float*) carve((size_t)BQ * DQ * 4);
  float* gamma  = (float*) carve((size_t)BQ * DQ * 4);
  float* beta   = (float*) carve((size_t)BQ * DQ * 4);
  __bf16* w1b   = (__bf16*)carve((size_t)PDQ * DQ * 2);
  __bf16* w2b   = (__bf16*)carve((size_t)DQ * PDQ * 2);
  __bf16* cb    = (__bf16*)carve((size_t)ROWS * DQ * 2);
  __bf16* sgb   = (__bf16*)carve((size_t)DQ * DQ * 2);
  __bf16* xfb   = (__bf16*)carve((size_t)ROWS * DQ * 2);
  __bf16* lmb   = (__bf16*)carve((size_t)VQ * DQ * 2);

  // one-time (per call) weight conversions
  k_cvt_bf16<<<dim3((unsigned)(((long long)DQ * DQ + 255) / 256)), 256, 0, stream>>>(sg, sgb, (long long)DQ * DQ);
  k_cvt_bf16<<<dim3((unsigned)(((long long)VQ * DQ + 255) / 256)), 256, 0, stream>>>(lm, lmb, (long long)VQ * DQ);

  k_embed<<<dim3(ROWS * DQ / 256), 256, 0, stream>>>(tokens, embed_w, x);

  for (int i = 0; i < NCQ; ++i) {
    int c = i % NSQ;
    hipMemcpyAsync(xn, x, (size_t)ROWS * DQ * 4, hipMemcpyDeviceToDevice, stream);
    k_rotate<<<dim3(ROWS * NPQ / 256), 256, 0, stream>>>(x, xn, angles + c * NPQ,
                                                         plane_i + c * NPQ, plane_j + c * NPQ);
    k_colmean<<<dim3(BQ * DQ / 256), 256, 0, stream>>>(xn, mbuf);
    k_router<<<dim3(BQ), 256, 0, stream>>>(mbuf, rw + (size_t)c * NMQ * DQ,
                                           msc + (size_t)c * NMQ * DQ,
                                           msh + (size_t)c * NMQ * DQ, gamma, beta);
    k_cvt_bf16<<<dim3((PDQ * DQ + 255) / 256), 256, 0, stream>>>(w1 + (size_t)c * PDQ * DQ, w1b, (long long)PDQ * DQ);
    k_cvt_bf16<<<dim3((DQ * PDQ + 255) / 256), 256, 0, stream>>>(w2 + (size_t)c * DQ * PDQ, w2b, (long long)DQ * PDQ);
    k_cycle_mlp<<<dim3(ROWS / 128), 256, 0, stream>>>(x, xn, gamma, beta, w1b, b1 + c * PDQ,
                                                      w2b, b2 + c * DQ, nw + c * DQ, csc, i, dlt);
    if (i % 7 == 6) {
      k_ema<<<dim3(BQ * DQ / 256), 256, 0, stream>>>(x, sd, causal);
      k_cvt_bf16<<<dim3((unsigned)(((long long)ROWS * DQ + 255) / 256)), 256, 0, stream>>>(causal, cb, (long long)ROWS * DQ);
      k_gemm<<<dim3(DQ / 128, ROWS / 128), 256, 0, stream>>>(cb, sgb, x, ROWS, DQ, DQ, 0);
    }
  }
  k_finalnorm<<<dim3(ROWS), 256, 0, stream>>>(x, onw, xfb);
  k_gemm<<<dim3((VQ + 127) / 128, ROWS / 128), 256, 0, stream>>>(xfb, lmb, out, ROWS, VQ, DQ, 1);
}
